// Classify_MLPPredictor_82592221102892
// MI455X (gfx1250) — compile-verified, matrix operand order checked by reading, not run on hardware
//
#include <hip/hip_runtime.h>

typedef __attribute__((ext_vector_type(16))) _Float16 v16h;
typedef __attribute__((ext_vector_type(8)))  _Float16 v8h;
typedef __attribute__((ext_vector_type(8)))  float    v8f;

#define IN_DIMS   128
#define N_CLASSES 128
// 256 halves + 8 halves (16B) pad per row: breaks the 512B-stride bank pattern
#define WT_STRIDE 264

// ---------------- prep: h (f32) -> h_f16 in workspace ----------------
__global__ void cvt_h_f16(const float* __restrict__ h,
                          _Float16* __restrict__ hf, int total) {
    int i = (blockIdx.x * blockDim.x + threadIdx.x) * 8;
    if (i + 8 <= total) {
        float4 a = *(const float4*)(h + i);
        float4 b = *(const float4*)(h + i + 4);
        v8h o;
        o[0] = (_Float16)a.x; o[1] = (_Float16)a.y;
        o[2] = (_Float16)a.z; o[3] = (_Float16)a.w;
        o[4] = (_Float16)b.x; o[5] = (_Float16)b.y;
        o[6] = (_Float16)b.z; o[7] = (_Float16)b.w;
        *(v8h*)(hf + i) = o;
    } else {
        for (; i < total; ++i) hf[i] = (_Float16)h[i];
    }
}

// ---------------- main: per-edge MLP via WMMA ----------------
// One wave handles TWO 16-edge M-tiles (32 edges): each B fragment from LDS
// feeds 2 WMMAs (ds:wmma = 1:1).
__global__ __launch_bounds__(256)
void gnn_edge_mlp_wmma(const _Float16* __restrict__ hf,
                       const int* __restrict__ src,
                       const int* __restrict__ dst,
                       const float* __restrict__ W,     // 256x128 row-major f32
                       const float* __restrict__ bias,  // 128 f32
                       float* __restrict__ out,         // n_edges x 128 f32
                       int n_edges, int n_pairs) {
    __shared__ _Float16 sWT[N_CLASSES * WT_STRIDE];   // W transposed, f16: sWT[n][k]

    // Stage W^T into LDS: each thread fills half of one n-row (contiguous LDS writes).
    {
        int n  = threadIdx.x >> 1;            // 0..127
        int k0 = (threadIdx.x & 1) * 128;     // 0 or 128
        #pragma unroll 4
        for (int kk = 0; kk < 128; ++kk) {
            int k = k0 + kk;
            sWT[n * WT_STRIDE + k] = (_Float16)W[k * N_CLASSES + n];
        }
    }
    __syncthreads();

    const int wave = threadIdx.x >> 5;
    const int lane = threadIdx.x & 31;
    const int m    = lane & 15;               // matrix row (A) / col (B) owned by lane
    const int sub  = (lane < 16) ? 0 : 8;     // K sub-offset per ISA 16-bit layout

    const int pair = blockIdx.x * 8 + wave;   // wave-uniform
    if (pair >= n_pairs) return;

    const int tile0 = pair * 2;
    const int last  = n_edges - 1;

    int e0 = tile0 * 16 + m;
    int e1 = e0 + 16;
    int ec0 = (e0 < n_edges) ? e0 : last;
    int ec1 = (e1 < n_edges) ? e1 : last;
    const _Float16* hs0 = hf + (size_t)src[ec0] * IN_DIMS;
    const _Float16* hd0 = hf + (size_t)dst[ec0] * IN_DIMS;
    const _Float16* hs1 = hf + (size_t)src[ec1] * IN_DIMS;
    const _Float16* hd1 = hf + (size_t)dst[ec1] * IN_DIMS;

    // Per-lane bias values (one per 16-wide N tile)
    float bv[8];
    #pragma unroll
    for (int nt = 0; nt < 8; ++nt) bv[nt] = bias[nt * 16 + m];

    v8f acc0[8], acc1[8];
    #pragma unroll
    for (int nt = 0; nt < 8; ++nt) {
        acc0[nt] = (v8f){0.f,0.f,0.f,0.f,0.f,0.f,0.f,0.f};
        acc1[nt] = (v8f){0.f,0.f,0.f,0.f,0.f,0.f,0.f,0.f};
    }

    // ---- K loop: per kt load 2 A fragments, stream 8 B fragments, 16 WMMAs ----
    // A layout (16-bit, 16x32): lane<16 : K = kt*32 + {0..7, 16..23};
    //                           lane>=16: K = kt*32 + {8..15, 24..31}
    #pragma unroll
    for (int kt = 0; kt < 8; ++kt) {
        const _Float16* b0 = (kt < 4) ? (hs0 + kt * 32) : (hd0 + (kt - 4) * 32);
        const _Float16* b1 = (kt < 4) ? (hs1 + kt * 32) : (hd1 + (kt - 4) * 32);
        v8h a0lo = *(const v8h*)(b0 + sub);
        v8h a0hi = *(const v8h*)(b0 + 16 + sub);
        v8h a1lo = *(const v8h*)(b1 + sub);
        v8h a1hi = *(const v8h*)(b1 + 16 + sub);
        v16h a0 = __builtin_shufflevector(a0lo, a0hi,
                      0,1,2,3,4,5,6,7,8,9,10,11,12,13,14,15);
        v16h a1 = __builtin_shufflevector(a1lo, a1hi,
                      0,1,2,3,4,5,6,7,8,9,10,11,12,13,14,15);

        #pragma unroll
        for (int nt = 0; nt < 8; ++nt) {
            const _Float16* wr = &sWT[(nt * 16 + m) * WT_STRIDE + kt * 32 + sub];
            v8h lo = *(const v8h*)wr;          // ds_load_b128
            v8h hi = *(const v8h*)(wr + 16);   // ds_load_b128
            v16h bf = __builtin_shufflevector(lo, hi,
                        0,1,2,3,4,5,6,7,8,9,10,11,12,13,14,15);
            acc0[nt] = __builtin_amdgcn_wmma_f32_16x16x32_f16(
                           false, a0, false, bf, (short)0, acc0[nt], false, false);
            acc1[nt] = __builtin_amdgcn_wmma_f32_16x16x32_f16(
                           false, a1, false, bf, (short)0, acc1[nt], false, false);
        }
    }

    // ---- epilogue: bias + sigmoid + store ----
    // C layout: VGPR v -> row M = v (lanes 0-15) / v+8 (lanes 16-31), col N = lane&15
    const int roff = (lane < 16) ? 0 : 8;
    #pragma unroll
    for (int t = 0; t < 2; ++t) {
        const v8f* acc = t ? acc1 : acc0;
        const int rbase = (tile0 + t) * 16 + roff;
        #pragma unroll
        for (int nt = 0; nt < 8; ++nt) {
            int col = nt * 16 + m;
            #pragma unroll
            for (int v = 0; v < 8; ++v) {
                int row = rbase + v;
                if (row < n_edges) {
                    float x  = acc[nt][v] + bv[nt];
                    float ex = __expf(-x);
                    out[(size_t)row * N_CLASSES + col] =
                        __builtin_amdgcn_rcpf(1.0f + ex);
                }
            }
        }
    }
}

extern "C" void kernel_launch(void* const* d_in, const int* in_sizes, int n_in,
                              void* d_out, int out_size, void* d_ws, size_t ws_size,
                              hipStream_t stream) {
    const float* h   = (const float*)d_in[0];
    const int*   src = (const int*)d_in[1];
    const int*   dst = (const int*)d_in[2];
    const float* W   = (const float*)d_in[3];
    const float* b   = (const float*)d_in[4];
    float*       out = (float*)d_out;

    int h_total = in_sizes[0];      // n_nodes * 128
    int n_edges = in_sizes[1];

    _Float16* hf = (_Float16*)d_ws; // 2 * h_total bytes of workspace

    int cvt_threads = (h_total + 7) / 8;
    int cvt_blocks  = (cvt_threads + 255) / 256;
    cvt_h_f16<<<cvt_blocks, 256, 0, stream>>>(h, hf, h_total);

    int n_tiles = (n_edges + 15) / 16;
    int n_pairs = (n_tiles + 1) / 2;     // each wave: 2 M-tiles (32 edges)
    int blocks  = (n_pairs + 7) / 8;     // 8 waves per block
    gnn_edge_mlp_wmma<<<blocks, 256, 0, stream>>>(hf, src, dst, W, b, out,
                                                  n_edges, n_pairs);
}
